// RNNBlock_40441412059392
// MI455X (gfx1250) — matrix operand order repaired and checked by reading, not compiled, not run on hardware
//
#include <hip/hip_runtime.h>
#include <stdint.h>

// ---------------------------------------------------------------------------
// MI455X (gfx1250) bidirectional RNN block.
// GEMMs: bf16 WMMA (v_wmma_f32_16x16x32_bf16), f32 accumulation,
//        async global->LDS tile streaming (ASYNCcnt) with double buffering.
//        Wave tile 64x64 (4x4 fragments) -> 1 ds_load_b128 per WMMA.
// Scans: channel-parallel sequential loops (memory-bound, tiny vs GEMMs).
// ---------------------------------------------------------------------------

#define BDIM 8
#define SDIM 4096
#define DDIM 1024
#define MROWS (BDIM * SDIM)          // 32768

typedef __attribute__((ext_vector_type(16))) __bf16 v16bf;
typedef __attribute__((ext_vector_type(8)))  float  v8f;

__device__ __forceinline__ uint16_t f2bf(float f) {
    uint32_t u = __builtin_bit_cast(uint32_t, f);
    uint32_t r = u + 0x7FFFu + ((u >> 16) & 1u);   // round-to-nearest-even
    return (uint16_t)(r >> 16);
}
__device__ __forceinline__ float bf2f(uint16_t h) {
    uint32_t u = ((uint32_t)h) << 16;
    return __builtin_bit_cast(float, u);
}
__device__ __forceinline__ float sigmoidf_(float x) {
    return 1.0f / (1.0f + expf(-x));
}

// ---------------------------------------------------------------------------
// Pointwise kernels
// ---------------------------------------------------------------------------

// f32 [K][N] weight -> bf16 [N][K] (transposed once; reused by all GEMM blocks)
__global__ void __launch_bounds__(256) transpose_cvt_bf16(const float* __restrict__ in,
                                                          uint16_t* __restrict__ out) {
    __shared__ float tile[32][33];
    int tx = threadIdx.x & 31;
    int ty = threadIdx.x >> 5;         // 0..7
    int n0 = blockIdx.x * 32;
    int k0 = blockIdx.y * 32;
#pragma unroll
    for (int i = 0; i < 32; i += 8)
        tile[ty + i][tx] = in[(size_t)(k0 + ty + i) * DDIM + n0 + tx];
    __syncthreads();
#pragma unroll
    for (int i = 0; i < 32; i += 8)
        out[(size_t)(n0 + ty + i) * DDIM + k0 + tx] = f2bf(tile[tx][ty + i]);
}

__global__ void bias_sum_k(const float* __restrict__ b0, const float* __restrict__ b1,
                           float* __restrict__ out) {
    int i = threadIdx.x;
    out[i] = b0[i] + b1[i];
}

__global__ void __launch_bounds__(256) gelu_bf16(const float* __restrict__ x,
                                                 uint16_t* __restrict__ out) {
    size_t i = (size_t)blockIdx.x * blockDim.x + threadIdx.x;
    float v = x[i];
    float t = 0.7978845608028654f * (v + 0.044715f * v * v * v);
    out[i] = f2bf(0.5f * v * (1.0f + tanhf(t)));
}

// RG-LRU gate math. a_out aliases gx_lin, s_out aliases ga_lin (same-index
// read-before-write within a thread: safe; no __restrict__ on those).
__global__ void __launch_bounds__(256) ew_gates(const uint16_t* __restrict__ v,
                                                const uint16_t* gx_lin,
                                                const uint16_t* ga_lin,
                                                const float* __restrict__ a_param,
                                                uint16_t* a_out, uint16_t* s_out) {
    size_t i = (size_t)blockIdx.x * blockDim.x + threadIdx.x;
    int d = (int)(i & (DDIM - 1));
    float ap = a_param[d];
    float sp = (ap > 20.0f) ? ap : log1pf(expf(ap));   // softplus
    float vv  = bf2f(v[i]);
    float gx  = sigmoidf_(bf2f(gx_lin[i]));
    float ga  = sigmoidf_(bf2f(ga_lin[i]));
    float log_a = -8.0f * ga * sp;
    float a  = expf(log_a);
    float sc = sqrtf(fmaxf(0.0f, 1.0f - expf(2.0f * log_a))) * gx * vv;
    a_out[i] = f2bf(a);
    s_out[i] = f2bf(sc);
}

// ---------------------------------------------------------------------------
// Scans: one thread per (b,d) channel; lanes cover consecutive d -> coalesced.
// ---------------------------------------------------------------------------

__global__ void __launch_bounds__(256) scan_fwd(const uint16_t* __restrict__ u,
                                                const float* __restrict__ a_fwd,
                                                uint16_t* __restrict__ hf) {
    int c = blockIdx.x * blockDim.x + threadIdx.x;   // 0 .. B*D-1
    int b = c >> 10;
    int d = c & (DDIM - 1);
    float a = a_fwd[d];
    size_t base = (size_t)b * SDIM * DDIM + d;
    float h = 0.0f;
#pragma unroll 4
    for (int s = 0; s < SDIM; ++s) {
        size_t idx = base + (size_t)s * DDIM;
        h = fmaf(a, h, bf2f(u[idx]));
        hf[idx] = f2bf(h);
    }
}

__global__ void __launch_bounds__(256) scan_bwd(const uint16_t* __restrict__ a,
                                                const uint16_t* __restrict__ sc,
                                                uint16_t* __restrict__ hb) {
    int c = blockIdx.x * blockDim.x + threadIdx.x;
    int b = c >> 10;
    int d = c & (DDIM - 1);
    size_t base = (size_t)b * SDIM * DDIM + d;
    float h = 0.0f;
#pragma unroll 4
    for (int s = SDIM - 1; s >= 0; --s) {
        size_t idx = base + (size_t)s * DDIM;
        h = fmaf(bf2f(a[idx]), h, bf2f(sc[idx]));
        hb[idx] = f2bf(h);
    }
}

// ---------------------------------------------------------------------------
// bf16 WMMA GEMM: C[M,N] = A[M,K] @ Wt[N,K]^T (+ epilogue)
//   MODE 0: bf16 out = acc + bias[col]
//   MODE 1: f32  out = acc + bias[col] + res[row,col]   (residual add)
//   MODE 2: f32  out += acc                              (accumulate pass)
// Block tile 128x256, BK=32. 8 waves: 2 along M x 4 along N, each wave 64x64
// = 4x4 fragments of v_wmma_f32_16x16x32_bf16 (16 WMMA per 16 ds_load_b128).
// Tiles stream global->LDS via GLOBAL_LOAD_ASYNC_TO_LDS_B128 (ASYNCcnt),
// double-buffered; both A and W tiles share the [row][K] layout.
// ---------------------------------------------------------------------------

#define BM 128
#define BN 256
#define BK 32
#define LDT 40                        // padded LDS row stride (elems); 80B rows
#define ROWS_TOT (BM + BN)            // 384 rows per tile-pair
#define BUF_ELEMS (ROWS_TOT * LDT)    // 15360 uint16 per buffer
#define BUF_BYTES (BUF_ELEMS * 2)     // 30720 bytes per buffer

union FragBF {
    uint4 u[2];
    v16bf v;
};

__device__ __forceinline__ void async_tile16(uint32_t lds_byte, uint64_t gaddr) {
    // per-lane: 16 bytes global -> LDS, tracked by ASYNCcnt (no VGPR round-trip)
    asm volatile("global_load_async_to_lds_b128 %0, %1, off"
                 :: "v"(lds_byte), "v"(gaddr)
                 : "memory");
}
__device__ __forceinline__ void wait_async0() {
    asm volatile("s_wait_asynccnt 0x0" ::: "memory");
}

template <int MODE>
__global__ void __launch_bounds__(256) gemm_bf16(const uint16_t* __restrict__ A,
                                                 const uint16_t* __restrict__ Wt,
                                                 const float* __restrict__ bias,
                                                 const float* __restrict__ res,
                                                 uint16_t* __restrict__ outb,
                                                 float* __restrict__ outf,
                                                 int M, int N, int K) {
    // single shared object -> LDS offset 0; async writes use raw byte offsets
    __shared__ __align__(16) uint16_t lds[2 * BUF_ELEMS];

    const int tid    = threadIdx.x;
    const int lane   = tid & 31;
    const int wave   = tid >> 5;          // 0..7
    const int waveM  = wave & 1;          // 2 waves along M
    const int waveN  = wave >> 1;         // 4 waves along N
    const int m_lane = lane & 15;
    const int kh     = lane >> 4;         // K-half select per WMMA layout

    const int blockM = blockIdx.x * BM;
    const int blockN = blockIdx.y * BN;

    // --- per-thread loader: 6 x 16B chunks (2 A rows, 4 W rows) ---
    const int col8 = (tid & 3) << 3;      // k-offset 0/8/16/24 elems
    const int r0   = tid >> 2;            // 0..63
    uint32_t la[6];
    uint64_t ga[6];
    la[0] = (uint32_t)(((r0      ) * LDT + col8) * 2);
    la[1] = (uint32_t)(((r0 +  64) * LDT + col8) * 2);
    ga[0] = (uint64_t)(uintptr_t)(A + (size_t)(blockM + r0     ) * K + col8);
    ga[1] = (uint64_t)(uintptr_t)(A + (size_t)(blockM + r0 + 64) * K + col8);
#pragma unroll
    for (int j = 0; j < 4; ++j) {
        int rb = r0 + j * 64;             // W-tile row 0..255
        la[2 + j] = (uint32_t)(((BM + rb) * LDT + col8) * 2);
        ga[2 + j] = (uint64_t)(uintptr_t)(Wt + (size_t)(blockN + rb) * K + col8);
    }

    v8f c[4][4];
#pragma unroll
    for (int mi = 0; mi < 4; ++mi)
#pragma unroll
        for (int ni = 0; ni < 4; ++ni)
            c[mi][ni] = (v8f){0.f, 0.f, 0.f, 0.f, 0.f, 0.f, 0.f, 0.f};

    // prologue: stream tile 0 into buffer 0
#pragma unroll
    for (int j = 0; j < 6; ++j) async_tile16(la[j], ga[j]);
#pragma unroll
    for (int j = 0; j < 6; ++j) ga[j] += BK * 2;

    const int NIT = K / BK;
    for (int it = 0; it < NIT; ++it) {
        wait_async0();          // this wave's tile chunks have landed in LDS
        __syncthreads();        // everyone's chunks landed / prev reads retired

        if (it + 1 < NIT) {     // stream next tile into the other buffer
            uint32_t boff = (uint32_t)(((it + 1) & 1) * BUF_BYTES);
#pragma unroll
            for (int j = 0; j < 6; ++j) async_tile16(la[j] + boff, ga[j]);
#pragma unroll
            for (int j = 0; j < 6; ++j) ga[j] += BK * 2;
        }

        const uint16_t* As = lds + (it & 1) * BUF_ELEMS;
        const uint16_t* Bs = As + BM * LDT;

        // --- fragment loads (match 16-bit A 16x32 / B 32x16 VGPR layouts) ---
        FragBF af[4], bfr[4];
#pragma unroll
        for (int mi = 0; mi < 4; ++mi) {
            int row = waveM * 64 + mi * 16 + m_lane;
            const uint16_t* p = &As[row * LDT + kh * 8];
            af[mi].u[0] = *(const uint4*)p;          // K 0..7   (+8 if kh)
            af[mi].u[1] = *(const uint4*)(p + 16);   // K 16..23 (+8 if kh)
        }
#pragma unroll
        for (int ni = 0; ni < 4; ++ni) {
            int nr = waveN * 64 + ni * 16 + m_lane;
            const uint16_t* p = &Bs[nr * LDT + kh * 8];
            bfr[ni].u[0] = *(const uint4*)p;         // K 0..7   (+8 if kh)
            bfr[ni].u[1] = *(const uint4*)(p + 16);  // K 16..23 (+8 if kh)
        }

#pragma unroll
        for (int mi = 0; mi < 4; ++mi)
#pragma unroll
            for (int ni = 0; ni < 4; ++ni)
                c[mi][ni] = __builtin_amdgcn_wmma_f32_16x16x32_bf16(
                    false, af[mi].v, false, bfr[ni].v,
                    (short)0, c[mi][ni], false, false);
    }

    // --- epilogue: C layout lanes 0-15 M=r, lanes 16-31 M=8+r; N=lane&15 ---
#pragma unroll
    for (int mi = 0; mi < 4; ++mi) {
#pragma unroll
        for (int ni = 0; ni < 4; ++ni) {
            int col  = blockN + waveN * 64 + ni * 16 + m_lane;
            int row0 = blockM + waveM * 64 + mi * 16 + kh * 8;
            float bv = (MODE == 2) ? 0.0f : bias[col];
            v8f acc = c[mi][ni];
#pragma unroll
            for (int r = 0; r < 8; ++r) {
                size_t idx = (size_t)(row0 + r) * N + col;
                if (MODE == 0)      outb[idx] = f2bf(acc[r] + bv);
                else if (MODE == 1) outf[idx] = acc[r] + bv + res[idx];
                else                outf[idx] += acc[r];
            }
        }
    }
}

// ---------------------------------------------------------------------------
// Host-side launch
// ---------------------------------------------------------------------------

extern "C" void kernel_launch(void* const* d_in, const int* in_sizes, int n_in,
                              void* d_out, int out_size, void* d_ws, size_t ws_size,
                              hipStream_t stream) {
    const float* x       = (const float*)d_in[0];
    const float* a_fwd   = (const float*)d_in[1];
    const float* Wf1     = (const float*)d_in[2];
    const float* bf1     = (const float*)d_in[3];
    const float* Wf2     = (const float*)d_in[4];
    const float* bf2     = (const float*)d_in[5];
    const float* Wbx     = (const float*)d_in[6];
    const float* bbx     = (const float*)d_in[7];
    const float* Wgx     = (const float*)d_in[8];
    const float* bgx     = (const float*)d_in[9];
    const float* Wga     = (const float*)d_in[10];
    const float* bga     = (const float*)d_in[11];
    const float* a_param = (const float*)d_in[12];
    const float* Wb2     = (const float*)d_in[13];
    const float* bb2     = (const float*)d_in[14];
    float* out = (float*)d_out;

    const size_t NE = (size_t)MROWS * DDIM;     // 33,554,432
    const int    D2 = DDIM * DDIM;              // 1,048,576

    uint16_t* slot0 = (uint16_t*)d_ws;          // xg -> gx_lin -> a
    uint16_t* slot1 = slot0 + NE;               // u  -> hb
    uint16_t* slot2 = slot1 + NE;               // v  -> hf
    uint16_t* slot3 = slot2 + NE;               // ga_lin -> scaled
    uint16_t* wb    = slot3 + NE;               // 6 bf16 transposed weights
    float* bias_sum = (float*)(wb + 6 * (size_t)D2);

    uint16_t* wbf1 = wb + 0 * (size_t)D2;
    uint16_t* wbf2 = wb + 1 * (size_t)D2;
    uint16_t* wbbx = wb + 2 * (size_t)D2;
    uint16_t* wbgx = wb + 3 * (size_t)D2;
    uint16_t* wbga = wb + 4 * (size_t)D2;
    uint16_t* wbb2 = wb + 5 * (size_t)D2;

    // 1) weight transpose+convert to bf16 [N][K]; combined output bias
    dim3 tgrid(DDIM / 32, DDIM / 32);
    transpose_cvt_bf16<<<tgrid, 256, 0, stream>>>(Wf1, wbf1);
    transpose_cvt_bf16<<<tgrid, 256, 0, stream>>>(Wf2, wbf2);
    transpose_cvt_bf16<<<tgrid, 256, 0, stream>>>(Wbx, wbbx);
    transpose_cvt_bf16<<<tgrid, 256, 0, stream>>>(Wgx, wbgx);
    transpose_cvt_bf16<<<tgrid, 256, 0, stream>>>(Wga, wbga);
    transpose_cvt_bf16<<<tgrid, 256, 0, stream>>>(Wb2, wbb2);
    bias_sum_k<<<1, DDIM, 0, stream>>>(bf2, bb2, bias_sum);

    // 2) xg = gelu(x)
    gelu_bf16<<<(int)(NE / 256), 256, 0, stream>>>(x, slot0);

    dim3 grid(MROWS / BM, DDIM / BN);   // (256, 4)

    // 3) u = xg @ Wf1 + bf1 ; v = xg @ Wbx + bbx
    gemm_bf16<0><<<grid, 256, 0, stream>>>(slot0, wbf1, bf1, nullptr, slot1, nullptr,
                                           MROWS, DDIM, DDIM);
    gemm_bf16<0><<<grid, 256, 0, stream>>>(slot0, wbbx, bbx, nullptr, slot2, nullptr,
                                           MROWS, DDIM, DDIM);
    // 4) gate linears from v
    gemm_bf16<0><<<grid, 256, 0, stream>>>(slot2, wbgx, bgx, nullptr, slot0, nullptr,
                                           MROWS, DDIM, DDIM);
    gemm_bf16<0><<<grid, 256, 0, stream>>>(slot2, wbga, bga, nullptr, slot3, nullptr,
                                           MROWS, DDIM, DDIM);
    // 5) RG-LRU gate math: a -> slot0, scaled -> slot3
    ew_gates<<<(int)(NE / 256), 256, 0, stream>>>(slot2, slot0, slot3, a_param,
                                                  slot0, slot3);
    // 6) scans: hf -> slot2, hb -> slot1
    scan_fwd<<<(BDIM * DDIM) / 256, 256, 0, stream>>>(slot1, a_fwd, slot2);
    scan_bwd<<<(BDIM * DDIM) / 256, 256, 0, stream>>>(slot0, slot3, slot1);
    // 7) out = hf @ Wf2 + (bf2+bb2) + x ; out += hb @ Wb2
    gemm_bf16<1><<<grid, 256, 0, stream>>>(slot2, wbf2, bias_sum, x, nullptr, out,
                                           MROWS, DDIM, DDIM);
    gemm_bf16<2><<<grid, 256, 0, stream>>>(slot1, wbb2, nullptr, nullptr, nullptr, out,
                                           MROWS, DDIM, DDIM);
}
